// Net_7241314861574
// MI455X (gfx1250) — compile-verified, matrix operand order checked by reading, not run on hardware
//
#include <hip/hip_runtime.h>
#include <hip/hip_bf16.h>
#include <math.h>
#include <stdint.h>

// ---------------------------------------------------------------------------
// Forward-Forward classifier inference for MI455X (gfx1250, wave32, WMMA).
// bf16 WMMA GEMMs, fp32 accumulation, deferred row-normalization fused into
// the epilogue (norm is a per-row scale => apply to the accumulator), async
// global->LDS staging (ASYNCcnt), deterministic per-block goodness partials.
// ---------------------------------------------------------------------------

typedef __attribute__((ext_vector_type(16))) __bf16 v16bf;
typedef __attribute__((ext_vector_type(8)))  __bf16 v8bf;
typedef __attribute__((ext_vector_type(8)))  float  v8f;

#define B_SZ   8192
#define NLAB   10
#define D_IN   784
#define D_PAD  800     // 784 padded to a multiple of 32 (zeros)
#define D_HID  2048
#define EPSV   1e-4f

// float -> bf16, round-to-nearest-even
__device__ __forceinline__ __bf16 f2bf(float f) {
  unsigned u = __float_as_uint(f);
  u = (u + 0x7FFFu + ((u >> 16) & 1u)) >> 16;
  unsigned short s = (unsigned short)u;
  __bf16 r;
  __builtin_memcpy(&r, &s, 2);
  return r;
}

__device__ __forceinline__ v16bf catbf(v8bf lo, v8bf hi) {
  union { v16bf v; v8bf h[2]; } u;
  u.h[0] = lo; u.h[1] = hi;
  return u.v;
}

// CDNA5 async copy: 16B per lane, global -> LDS, tracked by ASYNCcnt.
__device__ __forceinline__ void async_ld_b128(unsigned ldsOff, const void* g) {
  asm volatile("global_load_async_to_lds_b128 %0, %1, off"
               :: "v"(ldsOff), "v"(g) : "memory");
}
__device__ __forceinline__ void wait_async0() {
  asm volatile("s_wait_asynccnt 0x0" ::: "memory");
}

// ---------------------------------------------------------------------------
// Utility kernels
// ---------------------------------------------------------------------------

__global__ void zero_init_kernel(float* __restrict__ good, int n,
                                 unsigned* __restrict__ xmaxBits) {
  int i = blockIdx.x * blockDim.x + threadIdx.x;
  if (i == 0) *xmaxBits = 0u;
  for (; i < n; i += gridDim.x * blockDim.x) good[i] = 0.0f;
}

// Global max of x. x in [0,1) => non-negative => uint-bit compare is monotonic.
__global__ __launch_bounds__(256) void reduce_max_kernel(
    const float* __restrict__ x, int n, unsigned* __restrict__ xmaxBits) {
  __shared__ float sm[256];
  float m = 0.0f;
  for (int i = blockIdx.x * blockDim.x + threadIdx.x; i < n;
       i += gridDim.x * blockDim.x)
    m = fmaxf(m, x[i]);
  sm[threadIdx.x] = m;
  __syncthreads();
  for (int s = 128; s > 0; s >>= 1) {
    if (threadIdx.x < s) sm[threadIdx.x] = fmaxf(sm[threadIdx.x], sm[threadIdx.x + s]);
    __syncthreads();
  }
  if (threadIdx.x == 0) atomicMax(xmaxBits, __float_as_uint(sm[0]));
}

// fp32 [N][K] -> bf16 [N][Kpad] (zero-padded K tail)
__global__ void pack_weight_kernel(const float* __restrict__ W,
                                   __bf16* __restrict__ Wb,
                                   int N, int K, int Kpad) {
  long total = (long)N * Kpad;
  for (long i = blockIdx.x * (long)blockDim.x + threadIdx.x; i < total;
       i += (long)gridDim.x * blockDim.x) {
    int  k = (int)(i % Kpad);
    long n = i / Kpad;
    Wb[i] = f2bf((k < K) ? W[n * (long)K + k] : 0.0f);
  }
}

// Overlay label onto x: write UNNORMALIZED bf16 A0 [B][D_PAD] and the row
// scale inv[b] = 1/(||row||+eps). Normalization is applied in the GEMM.
__global__ __launch_bounds__(256) void overlay_kernel(
    const float* __restrict__ x, const unsigned* __restrict__ xmaxBits,
    __bf16* __restrict__ A0, float* __restrict__ inv, int label) {
  __shared__ float sm[256];
  const long  b    = blockIdx.x;
  const float* row = x + b * (long)D_IN;
  const float xmax = __uint_as_float(*xmaxBits);

  float ss = (threadIdx.x == 0) ? xmax * xmax : 0.0f;   // overlaid column l
  for (int k = threadIdx.x; k < D_IN; k += 256) {
    float v = (k < NLAB) ? 0.0f : row[k];
    ss += v * v;
  }
  sm[threadIdx.x] = ss;
  __syncthreads();
  for (int s = 128; s > 0; s >>= 1) {
    if (threadIdx.x < s) sm[threadIdx.x] += sm[threadIdx.x + s];
    __syncthreads();
  }
  if (threadIdx.x == 0) inv[b] = 1.0f / (sqrtf(sm[0]) + EPSV);

  for (int k = threadIdx.x; k < D_PAD; k += 256) {
    float v;
    if (k < NLAB)      v = (k == label) ? xmax : 0.0f;
    else if (k < D_IN) v = row[k];
    else               v = 0.0f;
    A0[b * (long)D_PAD + k] = f2bf(v);
  }
}

// Sum the 16 per-N-block partials (fixed order => deterministic),
// accumulate goodness, produce next layer's row scale.
__global__ void finalize_kernel(const float* __restrict__ partial,  // [16][B]
                                float* __restrict__ good,
                                float* __restrict__ inv) {
  int b = blockIdx.x * blockDim.x + threadIdx.x;
  if (b >= B_SZ) return;
  float s = 0.0f;
  for (int j = 0; j < 16; ++j) s += partial[(long)j * B_SZ + b];
  good[b] += s / (float)D_HID;
  inv[b] = 1.0f / (sqrtf(s) + EPSV);
}

__global__ void argmax_kernel(const float* __restrict__ good,
                              float* __restrict__ out) {
  int b = blockIdx.x * blockDim.x + threadIdx.x;
  if (b >= B_SZ) return;
  float best = good[b];
  int   bi   = 0;
  for (int l = 1; l < NLAB; ++l) {
    float v = good[(long)l * B_SZ + b];
    if (v > best) { best = v; bi = l; }   // strict > keeps earliest (jnp.argmax)
  }
  out[b] = (float)bi;
}

// ---------------------------------------------------------------------------
// Fused WMMA GEMM:
//   val = relu( (A[M][K] * W[N][K]^T) * inv[row] + bias[col] )
//   if writeH: Cout[row][col] = bf16(val)     (next layer's input)
//   partial[blockIdx.y][row] = sum_col_in_block val^2   (deterministic)
// Block tile 128x128, 8 wave32 waves, each wave 32(M) x 64(N) = 8 WMMA tiles.
// Double-buffered LDS filled by global_load_async_to_lds_b128 (ASYNCcnt).
// ---------------------------------------------------------------------------

#define TM 128
#define TN 128
#define TK 32
#define LDSLD 40   // padded row length in bf16 (80B) to spread LDS banks

__global__ __launch_bounds__(256) void gemm_fused_bf16(
    const __bf16* __restrict__ A, int lda,
    const __bf16* __restrict__ W, int ldw,
    const float* __restrict__ bias,
    const float* __restrict__ invp,
    __bf16* __restrict__ Cout, int ldc,
    float* __restrict__ partial,
    int K, int writeH) {
  __shared__ __align__(16) __bf16 As[2][TM * LDSLD];
  __shared__ __align__(16) __bf16 Bs[2][TN * LDSLD];
  __shared__ float invs[TM];
  __shared__ float rowsum[2][TM];

  const int tid  = threadIdx.x;
  const int lane = tid & 31;
  const int wave = tid >> 5;
  const int wm   = (wave & 3) * 32;   // wave M origin in block tile
  const int wn   = (wave >> 2) * 64;  // wave N origin in block tile
  const long bm  = (long)blockIdx.x * TM;
  const long bn  = (long)blockIdx.y * TN;

  // Async staging: each thread owns 2x16B of A and of W per K-step.
  const int gRow = tid >> 2;          // 0..63 (two passes cover 128 rows)
  const int gCol = (tid & 3) * 8;     // 0,8,16,24

  // Fragment addressing per the CDNA5 16-bit A/B striping.
  const int frow = lane & 15;
  const int fk   = (lane >> 4) * 8;

  v8f acc[2][4] = {};

  const __bf16* aPtr = A + (bm + gRow) * (long)lda + gCol;
  const __bf16* wPtr = W + (bn + gRow) * (long)ldw + gCol;

  // LDS byte offsets (low 32 bits of generic shared pointer = LDS offset).
  const unsigned asBase    = (unsigned)(uintptr_t)&As[0][0];
  const unsigned bsBase    = (unsigned)(uintptr_t)&Bs[0][0];
  const unsigned bufStride = (unsigned)(TM * LDSLD * sizeof(__bf16));
  const unsigned sOff  = (unsigned)((gRow * LDSLD + gCol) * sizeof(__bf16));
  const unsigned sOff2 = (unsigned)(((gRow + 64) * LDSLD + gCol) * sizeof(__bf16));

  // Issue tile 0.
  async_ld_b128(asBase + sOff,  aPtr);
  async_ld_b128(asBase + sOff2, aPtr + 64l * lda);
  async_ld_b128(bsBase + sOff,  wPtr);
  async_ld_b128(bsBase + sOff2, wPtr + 64l * ldw);
  if (tid < TM) invs[tid] = invp[bm + tid];
  wait_async0();
  __syncthreads();

  const int ksteps = K / TK;
  for (int ks = 0; ks < ksteps; ++ks) {
    const int  cur  = ks & 1;
    const bool more = (ks + 1) < ksteps;

    if (more) {  // fill the idle buffer asynchronously while we compute
      const unsigned nb = (unsigned)(cur ^ 1) * bufStride;
      const long     ko = (long)(ks + 1) * TK;
      async_ld_b128(asBase + nb + sOff,  aPtr + ko);
      async_ld_b128(asBase + nb + sOff2, aPtr + ko + 64l * lda);
      async_ld_b128(bsBase + nb + sOff,  wPtr + ko);
      async_ld_b128(bsBase + nb + sOff2, wPtr + ko + 64l * ldw);
    }

    v16bf afrag[2], wfrag[4];
#pragma unroll
    for (int mt = 0; mt < 2; ++mt) {
      const __bf16* p = &As[cur][(wm + mt * 16 + frow) * LDSLD + fk];
      afrag[mt] = catbf(*(const v8bf*)p, *(const v8bf*)(p + 16));
    }
#pragma unroll
    for (int nt = 0; nt < 4; ++nt) {
      const __bf16* p = &Bs[cur][(wn + nt * 16 + frow) * LDSLD + fk];
      wfrag[nt] = catbf(*(const v8bf*)p, *(const v8bf*)(p + 16));
    }

#pragma unroll
    for (int mt = 0; mt < 2; ++mt)
#pragma unroll
      for (int nt = 0; nt < 4; ++nt)
        acc[mt][nt] = __builtin_amdgcn_wmma_f32_16x16x32_bf16(
            false, afrag[mt], false, wfrag[nt], (short)0, acc[mt][nt],
            false, false);

    if (more) {
      wait_async0();     // our async writes to the idle buffer are done
      __syncthreads();   // everyone's are done; reads of it also retired
    }
  }

  // Epilogue. C/D layout: VGPR v -> M = (lane>>4)*8 + v, N = lane&15.
  float bv[4];
#pragma unroll
  for (int nt = 0; nt < 4; ++nt) bv[nt] = bias[bn + wn + nt * 16 + frow];

#pragma unroll
  for (int mt = 0; mt < 2; ++mt) {
#pragma unroll
    for (int v = 0; v < 8; ++v) {
      const int   rloc = wm + mt * 16 + (lane >> 4) * 8 + v;
      const float invr = invs[rloc];
      float ssv = 0.0f;
#pragma unroll
      for (int nt = 0; nt < 4; ++nt) {
        float val = acc[mt][nt][v] * invr + bv[nt];
        val = fmaxf(val, 0.0f);
        if (writeH) {
          const long col = bn + wn + nt * 16 + frow;
          Cout[(bm + rloc) * (long)ldc + col] = f2bf(val);
        }
        ssv += val * val;
      }
      // reduce across the 16 lanes sharing this row (xor stays in-half)
      ssv += __shfl_xor(ssv, 1, 32);
      ssv += __shfl_xor(ssv, 2, 32);
      ssv += __shfl_xor(ssv, 4, 32);
      ssv += __shfl_xor(ssv, 8, 32);
      if ((lane & 15) == 0) rowsum[wave >> 2][rloc] = ssv;
    }
  }
  __syncthreads();
  if (tid < TM)
    partial[(long)blockIdx.y * B_SZ + bm + tid] = rowsum[0][tid] + rowsum[1][tid];
}

// ---------------------------------------------------------------------------
// Host launch
// ---------------------------------------------------------------------------

extern "C" void kernel_launch(void* const* d_in, const int* in_sizes, int n_in,
                              void* d_out, int out_size, void* d_ws,
                              size_t ws_size, hipStream_t stream) {
  const float* x  = (const float*)d_in[0];
  const float* W1 = (const float*)d_in[1];
  const float* b1 = (const float*)d_in[2];
  const float* W2 = (const float*)d_in[3];
  const float* b2 = (const float*)d_in[4];
  const float* W3 = (const float*)d_in[5];
  const float* b3 = (const float*)d_in[6];

  char* ws = (char*)d_ws;
  size_t off = 0;
  unsigned* xmaxBits = (unsigned*)(ws + off); off += 256;
  __bf16* W1b = (__bf16*)(ws + off); off += (size_t)D_HID * D_PAD * 2;
  __bf16* W2b = (__bf16*)(ws + off); off += (size_t)D_HID * D_HID * 2;
  __bf16* W3b = (__bf16*)(ws + off); off += (size_t)D_HID * D_HID * 2;
  float*  good = (float*)(ws + off); off += (size_t)NLAB * B_SZ * 4;
  float*  invB = (float*)(ws + off); off += (size_t)B_SZ * 4;
  float*  part = (float*)(ws + off); off += (size_t)16 * B_SZ * 4;
  __bf16* Aa = (__bf16*)(ws + off); off += (size_t)B_SZ * D_HID * 2;
  __bf16* Ab = (__bf16*)(ws + off); off += (size_t)B_SZ * D_HID * 2;

  zero_init_kernel<<<256, 256, 0, stream>>>(good, NLAB * B_SZ, xmaxBits);
  reduce_max_kernel<<<512, 256, 0, stream>>>(x, B_SZ * D_IN, xmaxBits);
  pack_weight_kernel<<<1024, 256, 0, stream>>>(W1, W1b, D_HID, D_IN, D_PAD);
  pack_weight_kernel<<<2048, 256, 0, stream>>>(W2, W2b, D_HID, D_HID, D_HID);
  pack_weight_kernel<<<2048, 256, 0, stream>>>(W3, W3b, D_HID, D_HID, D_HID);

  const dim3 gg(B_SZ / TM, D_HID / TN);
  const int fin_blocks = (B_SZ + 255) / 256;
  for (int l = 0; l < NLAB; ++l) {
    float* gl = good + (size_t)l * B_SZ;
    overlay_kernel<<<B_SZ, 256, 0, stream>>>(x, xmaxBits, Aa, invB, l);
    gemm_fused_bf16<<<gg, 256, 0, stream>>>(Aa, D_PAD, W1b, D_PAD, b1, invB,
                                            Ab, D_HID, part, D_PAD, 1);
    finalize_kernel<<<fin_blocks, 256, 0, stream>>>(part, gl, invB);
    gemm_fused_bf16<<<gg, 256, 0, stream>>>(Ab, D_HID, W2b, D_HID, b2, invB,
                                            Aa, D_HID, part, D_HID, 1);
    finalize_kernel<<<fin_blocks, 256, 0, stream>>>(part, gl, invB);
    gemm_fused_bf16<<<gg, 256, 0, stream>>>(Aa, D_HID, W3b, D_HID, b3, invB,
                                            Ab, D_HID, part, D_HID, 0);
    finalize_kernel<<<fin_blocks, 256, 0, stream>>>(part, gl, invB);
  }
  argmax_kernel<<<(B_SZ + 255) / 256, 256, 0, stream>>>(good, (float*)d_out);

  (void)in_sizes; (void)n_in; (void)out_size; (void)ws_size;
}
